// GatedCrossScaleBlock_66056597013043
// MI455X (gfx1250) — matrix operand order repaired and checked by reading, compiled, not run on hardware
//
#include <hip/hip_runtime.h>
#include <hip/hip_bf16.h>
#include <math.h>

typedef __attribute__((ext_vector_type(16))) __bf16 v16bf;
typedef __attribute__((ext_vector_type(8)))  float  v8f;

#define C_CH 64
#define D_D  48
#define H_H  96
#define W_W  96
#define HW   (96*96)         // 9216
#define DHW  (48*96*96)      // 442368
#define NV   (2*DHW)         // 884736

// conv tile: 4x4x16 outputs, haloed Z-column set 6x6x18
#define TD 4
#define TH 4
#define TW 16
#define NZ  (6*6*18)         // 648 Z columns
#define NZT ((NZ + 15) / 16) // 41 N-tiles of 16 columns
#define ZSTRIDE 33           // 32 taps + 1 pad (bank-conflict free)

// ---------------------------------------------------------------------------
// Kernel A: tap-GEMM conv (WMMA bf16) + sigmoid spatial gate + GAP partials
// ---------------------------------------------------------------------------
__global__ void __launch_bounds__(256)
conv_gate_kernel(const float* __restrict__ skip,
                 const float* __restrict__ conv_w,
                 const float* __restrict__ conv_b,
                 float* __restrict__ spa_ws,
                 float* __restrict__ gap_ws)
{
  extern __shared__ float smem[];
  float* wLds = smem;                   // 1728 floats: conv_w staged
  float* Zl   = smem + 1728;            // NZ * ZSTRIDE floats: Z tiles
  float* spaL = Zl + NZ * ZSTRIDE;      // 256 floats: spa per output voxel

  const int tid  = threadIdx.x;
  const int lane = tid & 31;
  const int wave = tid >> 5;

  // stage weights: conv_w[0,c,kd,kh,kw] flat = c*27 + t,  t = kd*9+kh*3+kw
  for (int i = tid; i < 1728; i += 256) wLds[i] = conv_w[i];
  __syncthreads();

  // block -> tile origin
  int bid = blockIdx.x;
  int tw  = bid % 6;
  int th  = (bid / 6) % 24;
  int td  = (bid / 144) % 12;
  int b   = bid / 1728;
  int d0 = td * TD, h0 = th * TH, w0 = tw * TW;

  // -------- build A fragments: A = Wt [32 taps(M) x 64 ch(K)], bf16 ------
  // A 16x32, contiguous k (this code's order): lanes0-15 M=0..15 hold K=0..15
  //                        lanes16-31 hold K=16..31
  const int mrow  = lane & 15;
  const int abase = (lane < 16) ? 0 : 16;
  const int t1    = 16 + mrow;                 // taps 16..31 (>=27 zero-padded)
  v16bf a00, a01, a10, a11;                    // a{mtile}{kstep}
#pragma unroll
  for (int j = 0; j < 8; ++j) {
    int kj = abase + 2 * j;
#pragma unroll
    for (int e = 0; e < 2; ++e) {
      int idx = 2 * j + e;
      int cA = kj + e;          // K-step 0: channels 0..31
      int cB = 32 + kj + e;     // K-step 1: channels 32..63
      a00[idx] = (__bf16)wLds[cA * 27 + mrow];
      a01[idx] = (__bf16)wLds[cB * 27 + mrow];
      a10[idx] = (__bf16)((t1 < 27) ? wLds[cA * 27 + t1] : 0.f);
      a11[idx] = (__bf16)((t1 < 27) ? wLds[cB * 27 + t1] : 0.f);
    }
  }

  // -------- phase 1: Z[t, col] = Wt x Skip over haloed column set --------
  const float* skipB = skip + (size_t)b * C_CH * DHW;
  const int kbase = (lane < 16) ? 0 : 16;   // B 32x16: lanes0-15 K=0..15
  for (int t = wave; t < NZT; t += 8) {
    int zc  = t * 16 + (lane & 15);
    int zcc = (zc < NZ) ? zc : (NZ - 1);
    int zw = zcc % 18;
    int zh = (zcc / 18) % 6;
    int zd = zcc / 108;
    int d = d0 - 1 + zd, h = h0 - 1 + zh, w = w0 - 1 + zw;
    bool inb = (d >= 0) && (d < D_D) && (h >= 0) && (h < H_H) &&
               (w >= 0) && (w < W_W);
    // branchless zero-padding: clamp address, scale by mask
    int   voff = inb ? (d * HW + h * W_W + w) : 0;
    float msk  = inb ? 1.f : 0.f;
    const float* sp = skipB + voff;

    // burst-load all 32 channel samples for this lane's column (one wait),
    // then convert and run the 4 WMMAs back-to-back
    float vv[32];
#pragma unroll
    for (int q = 0; q < 16; ++q) vv[q]      = sp[(size_t)(kbase + q) * DHW];
#pragma unroll
    for (int q = 0; q < 16; ++q) vv[16 + q] = sp[(size_t)(32 + kbase + q) * DHW];

    v16bf bf0, bf1;
#pragma unroll
    for (int q = 0; q < 16; ++q) {
      bf0[q] = (__bf16)(vv[q] * msk);
      bf1[q] = (__bf16)(vv[16 + q] * msk);
    }

    v8f acc0 = {}; v8f acc1 = {};
    acc0 = __builtin_amdgcn_wmma_f32_16x16x32_bf16(false, a00, false, bf0,
                                                   (short)0, acc0, false, false);
    acc1 = __builtin_amdgcn_wmma_f32_16x16x32_bf16(false, a10, false, bf0,
                                                   (short)0, acc1, false, false);
    acc0 = __builtin_amdgcn_wmma_f32_16x16x32_bf16(false, a01, false, bf1,
                                                   (short)0, acc0, false, false);
    acc1 = __builtin_amdgcn_wmma_f32_16x16x32_bf16(false, a11, false, bf1,
                                                   (short)0, acc1, false, false);

    // D layout: lane l, elem r -> M = r + 8*(l>>4)
    if (zc < NZ) {
      int hf = lane >> 4;
      float* zp = Zl + zc * ZSTRIDE;
#pragma unroll
      for (int r = 0; r < 8; ++r) {
        zp[hf * 8 + r]      = acc0[r];
        zp[16 + hf * 8 + r] = acc1[r];
      }
    }
  }
  __syncthreads();

  // -------- phase 2a: gather 27 taps, sigmoid, emit spa ------------------
  int ow = tid & 15, oh = (tid >> 4) & 3, od = tid >> 6;
  float conv = 0.f;
#pragma unroll
  for (int kd = 0; kd < 3; ++kd)
#pragma unroll
    for (int kh = 0; kh < 3; ++kh)
#pragma unroll
      for (int kw = 0; kw < 3; ++kw) {
        int zcol = ((od + kd) * 6 + (oh + kh)) * 18 + (ow + kw);
        conv += Zl[zcol * ZSTRIDE + (kd * 9 + kh * 3 + kw)];
      }
  float spav = 1.f / (1.f + __expf(-(conv + conv_b[0])));
  {
    int d = d0 + od, h = h0 + oh, w = w0 + ow;
    int vox = ((b * D_D + d) * H_H + h) * W_W + w;
    spa_ws[vox] = spav;
  }
  spaL[tid] = spav;
  __syncthreads();

  // -------- phase 2b: GAP partials (skip tile is L2-hot) -----------------
#pragma unroll 1
  for (int c8 = 0; c8 < 8; ++c8) {
    int c = wave * 8 + c8;
    const float* sc = skipB + (size_t)c * DHW + d0 * HW + h0 * W_W + w0;
    float s = 0.f;
#pragma unroll
    for (int i0 = 0; i0 < 8; ++i0) {
      int i = i0 * 32 + lane;
      int iw = i & 15, ih = (i >> 4) & 3, id = i >> 6;
      s += sc[id * HW + ih * W_W + iw] * spaL[i];
    }
#pragma unroll
    for (int off = 16; off > 0; off >>= 1) s += __shfl_xor(s, off, 32);
    if (lane == 0) atomicAdd(&gap_ws[b * 64 + c], s);
  }
}

// ---------------------------------------------------------------------------
// Kernel B: tiny channel-gate MLP (one block)
// ---------------------------------------------------------------------------
__global__ void __launch_bounds__(128)
mlp_gate_kernel(const float* __restrict__ w1, const float* __restrict__ b1,
                const float* __restrict__ w2, const float* __restrict__ b2,
                const float* __restrict__ gap_ws, float* __restrict__ gate_ws)
{
  __shared__ float hL[2][16];
  int tid = threadIdx.x;
  if (tid < 32) {
    int j = tid & 15, bb = tid >> 4;
    float acc = b1[j];
    const float* g = gap_ws + bb * 64;
    for (int c = 0; c < 64; ++c)
      acc += w1[j * 64 + c] * (g[c] * (1.f / (float)DHW));
    hL[bb][j] = fmaxf(acc, 0.f);
  }
  __syncthreads();
  int c = tid & 63, bb = tid >> 6;
  float acc = b2[c];
#pragma unroll
  for (int j = 0; j < 16; ++j) acc += w2[c * 16 + j] * hL[bb][j];
  gate_ws[bb * 64 + c] = 1.f / (1.f + __expf(-acc));
}

// ---------------------------------------------------------------------------
// Kernel C: residual fuse + per-voxel LayerNorm over C=64
// ---------------------------------------------------------------------------
__global__ void __launch_bounds__(256)
fuse_ln_kernel(const float* __restrict__ dec_x, const float* __restrict__ skip,
               const float* __restrict__ spa_ws, const float* __restrict__ gate_ws,
               const float* __restrict__ ln_g, const float* __restrict__ ln_b,
               float* __restrict__ out)
{
  __shared__ float gateL[128], gL[64], bL[64];
  int tid = threadIdx.x;
  if (tid < 128) gateL[tid] = gate_ws[tid];
  if (tid < 64) { gL[tid] = ln_g[tid]; bL[tid] = ln_b[tid]; }
  __syncthreads();

  int v = blockIdx.x * 256 + tid;        // linear over B*D*H*W
  int bb = v / DHW;
  int voxoff = v - bb * DHW;
  float spav = spa_ws[v];
  size_t base = (size_t)bb * C_CH * DHW + (size_t)voxoff;

  float xs[64];
  float sum = 0.f, sum2 = 0.f;
#pragma unroll
  for (int c = 0; c < 64; ++c) {
    size_t idx = base + (size_t)c * DHW;
    float x = dec_x[idx] + skip[idx] * spav * gateL[bb * 64 + c];
    xs[c] = x; sum += x; sum2 += x * x;
  }
  float mu  = sum * (1.f / 64.f);
  float var = sum2 * (1.f / 64.f) - mu * mu;
  float rs  = rsqrtf(var + 1e-5f);
#pragma unroll
  for (int c = 0; c < 64; ++c) {
    size_t idx = base + (size_t)c * DHW;
    out[idx] = (xs[c] - mu) * rs * gL[c] + bL[c];
  }
}

// ---------------------------------------------------------------------------
extern "C" void kernel_launch(void* const* d_in, const int* in_sizes, int n_in,
                              void* d_out, int out_size, void* d_ws, size_t ws_size,
                              hipStream_t stream)
{
  (void)in_sizes; (void)n_in; (void)out_size; (void)ws_size;
  const float* dec_x  = (const float*)d_in[0];
  const float* skip   = (const float*)d_in[1];
  const float* conv_w = (const float*)d_in[2];
  const float* conv_b = (const float*)d_in[3];
  const float* w1     = (const float*)d_in[4];
  const float* b1     = (const float*)d_in[5];
  const float* w2     = (const float*)d_in[6];
  const float* b2     = (const float*)d_in[7];
  const float* ln_g   = (const float*)d_in[8];
  const float* ln_b   = (const float*)d_in[9];
  float* out = (float*)d_out;

  // workspace layout: spa [NV] | gap [128] | gate [128]   (~3.55 MB)
  float* spa_ws  = (float*)d_ws;
  float* gap_ws  = spa_ws + NV;
  float* gate_ws = gap_ws + 128;

  hipMemsetAsync(gap_ws, 0, 128 * sizeof(float), stream);

  size_t shmemA = (size_t)(1728 + NZ * ZSTRIDE + 256) * sizeof(float);
  conv_gate_kernel<<<3456, 256, shmemA, stream>>>(skip, conv_w, conv_b,
                                                  spa_ws, gap_ws);
  mlp_gate_kernel<<<1, 128, 0, stream>>>(w1, b1, w2, b2, gap_ws, gate_ws);
  fuse_ln_kernel<<<3456, 256, 0, stream>>>(dec_x, skip, spa_ws, gate_ws,
                                           ln_g, ln_b, out);
}
